// LandmarksFrontend_60413009985598
// MI455X (gfx1250) — matrix operand order repaired and checked by reading, not compiled, not run on hardware
//
#include <hip/hip_runtime.h>

#define Bx 8
#define Sx 256
#define Lnodes 68
#define DM 512
#define NH 8
#define DH 64
#define DFF 2048
#define NL 6
#define TOK 2048            // S*B tokens (also B*S graphs)
#define EMAX 512

typedef unsigned short u16;
typedef __attribute__((ext_vector_type(16))) __bf16 v16bf;
typedef __attribute__((ext_vector_type(8)))  float  v8f;

union BfFrag { v16bf v; u16 u[16]; uint4 q[2]; };

__device__ __forceinline__ u16 f2bf(float f) {
  unsigned int u = __float_as_uint(f);
  unsigned int r = (u + 0x7FFFu + ((u >> 16) & 1u)) >> 16;
  return (u16)r;
}

// ---------------------------------------------------------------------------
// fp32 -> bf16 conversion (weights)
// ---------------------------------------------------------------------------
__global__ void f32_to_bf16_kernel(const float* __restrict__ in, u16* __restrict__ out, long long n) {
  long long i = (long long)blockIdx.x * blockDim.x + threadIdx.x;
  long long stride = (long long)gridDim.x * blockDim.x;
  for (; i < n; i += stride) out[i] = f2bf(in[i]);
}

// ---------------------------------------------------------------------------
// GAT frontend: two GATConv layers per graph (2048 graphs), output bf16
// [G, 68*64] feeding the FC GEMM.
// ---------------------------------------------------------------------------
__global__ __launch_bounds__(128) void gat_kernel(
    const float* __restrict__ lm, const int* __restrict__ eidx, int E,
    const float* __restrict__ W1, const float* __restrict__ as1,
    const float* __restrict__ ad1, const float* __restrict__ b1,
    const float* __restrict__ W2, const float* __restrict__ as2,
    const float* __restrict__ ad2, const float* __restrict__ b2,
    u16* __restrict__ out_bf) {
  const int g = blockIdx.x;
  const int tid = threadIdx.x;
  __shared__ float sx[Lnodes * 2];
  __shared__ float h1[Lnodes * 16];
  __shared__ float o1[Lnodes * 16];
  __shared__ float h2[Lnodes * 64];
  __shared__ float esrc[Lnodes], edst[Lnodes], ssum[Lnodes];
  __shared__ float we[EMAX];
  __shared__ float W2s[64 * 16];
  __shared__ int ssrc[EMAX], sdst[EMAX];

  const int Ec = (E > EMAX) ? EMAX : E;
  for (int i = tid; i < Lnodes * 2; i += 128) sx[i] = lm[(size_t)g * Lnodes * 2 + i];
  for (int i = tid; i < Ec; i += 128) { ssrc[i] = eidx[i]; sdst[i] = eidx[E + i]; }
  for (int i = tid; i < 64 * 16; i += 128) W2s[i] = W2[i];
  __syncthreads();

  // ---- layer 1: linear (2 -> 16)
  for (int i = tid; i < Lnodes * 16; i += 128) {
    int n = i >> 4, f = i & 15;
    h1[i] = W1[f * 2] * sx[n * 2] + W1[f * 2 + 1] * sx[n * 2 + 1];
  }
  __syncthreads();
  for (int n = tid; n < Lnodes; n += 128) {
    float s = 0.f, d = 0.f;
    for (int f = 0; f < 16; ++f) { s += h1[n * 16 + f] * as1[f]; d += h1[n * 16 + f] * ad1[f]; }
    esrc[n] = s; edst[n] = d;
  }
  __syncthreads();
  for (int t = tid; t < Lnodes; t += 128) {
    float m = -1e30f;
    for (int e = 0; e < Ec; ++e)
      if (sdst[e] == t) { float v = esrc[ssrc[e]] + edst[t]; v = (v >= 0.f) ? v : 0.2f * v; m = fmaxf(m, v); }
    float sum = 0.f;
    for (int e = 0; e < Ec; ++e)
      if (sdst[e] == t) {
        float v = esrc[ssrc[e]] + edst[t]; v = (v >= 0.f) ? v : 0.2f * v;
        float ex = __expf(v - m); we[e] = ex; sum += ex;
      }
    ssum[t] = sum;
  }
  __syncthreads();
  for (int i = tid; i < Lnodes * 16; i += 128) {
    int t = i >> 4, f = i & 15;
    float acc = 0.f;
    for (int e = 0; e < Ec; ++e)
      if (sdst[e] == t) acc += we[e] * h1[ssrc[e] * 16 + f];
    float v = acc / ssum[t] + b1[f];
    o1[i] = (v > 0.f) ? v : 0.f;   // relu after gat_conv
  }
  __syncthreads();

  // ---- layer 2: linear (16 -> 64)
  for (int i = tid; i < Lnodes * 64; i += 128) {
    int n = i >> 6, f = i & 63;
    float acc = 0.f;
#pragma unroll
    for (int k = 0; k < 16; ++k) acc += W2s[f * 16 + k] * o1[n * 16 + k];
    h2[i] = acc;
  }
  __syncthreads();
  for (int n = tid; n < Lnodes; n += 128) {
    float s = 0.f, d = 0.f;
    for (int f = 0; f < 64; ++f) { s += h2[n * 64 + f] * as2[f]; d += h2[n * 64 + f] * ad2[f]; }
    esrc[n] = s; edst[n] = d;
  }
  __syncthreads();
  for (int t = tid; t < Lnodes; t += 128) {
    float m = -1e30f;
    for (int e = 0; e < Ec; ++e)
      if (sdst[e] == t) { float v = esrc[ssrc[e]] + edst[t]; v = (v >= 0.f) ? v : 0.2f * v; m = fmaxf(m, v); }
    float sum = 0.f;
    for (int e = 0; e < Ec; ++e)
      if (sdst[e] == t) {
        float v = esrc[ssrc[e]] + edst[t]; v = (v >= 0.f) ? v : 0.2f * v;
        float ex = __expf(v - m); we[e] = ex; sum += ex;
      }
    ssum[t] = sum;
  }
  __syncthreads();
  for (int i = tid; i < Lnodes * 64; i += 128) {
    int t = i >> 6, f = i & 63;
    float acc = 0.f;
    for (int e = 0; e < Ec; ++e)
      if (sdst[e] == t) acc += we[e] * h2[ssrc[e] * 64 + f];
    float v = acc / ssum[t] + b2[f];
    out_bf[(size_t)g * (Lnodes * 64) + i] = f2bf((v > 0.f) ? v : 0.f);
  }
}

// ---------------------------------------------------------------------------
// Batched bf16 WMMA GEMM, one wave computes a 32x32 tile of
//   C = act( scale * A x B^T + bias + residual )
// A: [M,K] bf16 row-major (row stride lda, 16B-aligned rows).
// B: [N,K] bf16 row-major (row stride ldb) -- i.e. x @ W.T with W=[N,K].
// Batch z decomposed as z = zo*inner_n + zi with per-level base offsets.
// 4 accumulators / 4 v_wmma per 32-wide K step; fragment loads are b128.
// Wave32 VGPR layouts per CDNA5 ISA 7.12.2.
// ---------------------------------------------------------------------------
__global__ __launch_bounds__(32) void wmma_gemm_bf16(
    const u16* __restrict__ A, long long lda, long long aoo, long long aoi,
    const u16* __restrict__ Bm, long long ldb, long long boo, long long boi,
    float* __restrict__ Cf, u16* __restrict__ Cb,
    long long ldc, long long coo, long long coi,
    const float* __restrict__ bias, const float* __restrict__ resid,
    int M, int N, int Kd, int inner_n, float scale, int relu) {
  const int tile_n = blockIdx.x;
  const int tile_m = blockIdx.y;
  const int z = blockIdx.z;
  const int zo = z / inner_n;
  const int zi = z - zo * inner_n;

  const long long cbase = (long long)zo * coo + (long long)zi * coi;

  const int lane = threadIdx.x & 31;
  const int lr = lane & 15;       // A: row m; B: row n; C: col n
  const int hi = lane >> 4;
  const int kbA = hi * 8;         // A half: K base 0 / 8 (+16 for upper run)
  const int kbB = hi * 8;         // B half: K base 0 / 8 (+16 for upper run)
  const int m0 = tile_m * 32, n0 = tile_n * 32;

  const u16* arow0 = A + (long long)zo * aoo + (long long)zi * aoi + (long long)(m0 + lr) * lda + kbA;
  const u16* arow1 = arow0 + 16 * lda;
  const u16* brow0 = Bm + (long long)zo * boo + (long long)zi * boi + (long long)(n0 + lr) * ldb + kbB;
  const u16* brow1 = brow0 + 16 * ldb;

  v8f acc00 = {}, acc01 = {}, acc10 = {}, acc11 = {};

  for (int k0 = 0; k0 < Kd; k0 += 32) {
    BfFrag a0, a1, b0, b1;
    a0.q[0] = *(const uint4*)(arow0);
    a0.q[1] = *(const uint4*)(arow0 + 16);
    a1.q[0] = *(const uint4*)(arow1);
    a1.q[1] = *(const uint4*)(arow1 + 16);
    b0.q[0] = *(const uint4*)(brow0);
    b0.q[1] = *(const uint4*)(brow0 + 16);
    b1.q[0] = *(const uint4*)(brow1);
    b1.q[1] = *(const uint4*)(brow1 + 16);
    arow0 += 32; arow1 += 32; brow0 += 32; brow1 += 32;
    acc00 = __builtin_amdgcn_wmma_f32_16x16x32_bf16(false, a0.v, false, b0.v, (short)0, acc00, false, false);
    acc01 = __builtin_amdgcn_wmma_f32_16x16x32_bf16(false, a0.v, false, b1.v, (short)0, acc01, false, false);
    acc10 = __builtin_amdgcn_wmma_f32_16x16x32_bf16(false, a1.v, false, b0.v, (short)0, acc10, false, false);
    acc11 = __builtin_amdgcn_wmma_f32_16x16x32_bf16(false, a1.v, false, b1.v, (short)0, acc11, false, false);
  }

  const v8f* accs[4] = { &acc00, &acc01, &acc10, &acc11 };
#pragma unroll
  for (int s = 0; s < 4; ++s) {
    const int sm = s >> 1, sn = s & 1;
    const int cn = n0 + sn * 16 + lr;
    const float bv = bias ? bias[cn] : 0.0f;
    const v8f a = *accs[s];
#pragma unroll
    for (int r = 0; r < 8; ++r) {
      int cm = m0 + sm * 16 + r + hi * 8;
      long long ci = cbase + (long long)cm * ldc + cn;
      float v = a[r] * scale + bv;
      if (resid) v += resid[ci];
      if (relu) v = (v > 0.f) ? v : 0.f;
      if (Cf) Cf[ci] = v;
      if (Cb) Cb[ci] = f2bf(v);
    }
  }
}

// ---------------------------------------------------------------------------
// transpose (b,s)->(s,b) + positional encoding; fp32 + bf16 outputs
// ---------------------------------------------------------------------------
__global__ __launch_bounds__(256) void posenc_kernel(
    const float* __restrict__ fc_out, const float* __restrict__ pe,
    float* __restrict__ x, u16* __restrict__ xb) {
  const int blk = blockIdx.x;       // b*S + s
  const int b = blk >> 8;
  const int s = blk & 255;
  const int orow = s * Bx + b;
  for (int c = threadIdx.x; c < DM; c += 256) {
    float v = fc_out[(size_t)blk * DM + c] + pe[(size_t)s * DM + c];
    x[(size_t)orow * DM + c] = v;
    xb[(size_t)orow * DM + c] = f2bf(v);
  }
}

// ---------------------------------------------------------------------------
// V transpose: vt[(b,h)][d][t] = qkv[(t*B+b)*1536 + 1024 + h*64 + d]
// makes attn@V use the [N,K] fast GEMM path.
// ---------------------------------------------------------------------------
__global__ __launch_bounds__(256) void transpose_v_kernel(
    const u16* __restrict__ qkv, u16* __restrict__ vt) {
  const int z = blockIdx.x;         // b*NH + h
  const int b = z >> 3;
  const int h = z & 7;
  for (int i = threadIdx.x; i < DH * Sx; i += 256) {
    int t = i >> 6;                 // 0..255
    int d = i & 63;                 // 0..63
    vt[((size_t)z * DH + d) * Sx + t] =
        qkv[((size_t)t * Bx + b) * (3 * DM) + 2 * DM + h * DH + d];
  }
}

// ---------------------------------------------------------------------------
// row softmax over 256 cols, output bf16
// ---------------------------------------------------------------------------
__global__ __launch_bounds__(256) void softmax_kernel(
    const float* __restrict__ sc, u16* __restrict__ at) {
  const long long base = (long long)blockIdx.x * 256;
  const int tid = threadIdx.x;
  __shared__ float red[256];
  float v = sc[base + tid];
  red[tid] = v; __syncthreads();
  for (int s = 128; s > 0; s >>= 1) { if (tid < s) red[tid] = fmaxf(red[tid], red[tid + s]); __syncthreads(); }
  float m = red[0]; __syncthreads();
  float e = __expf(v - m);
  red[tid] = e; __syncthreads();
  for (int s = 128; s > 0; s >>= 1) { if (tid < s) red[tid] += red[tid + s]; __syncthreads(); }
  at[base + tid] = f2bf(e / red[0]);
}

// ---------------------------------------------------------------------------
// LayerNorm over 512 features; writes fp32 + bf16
// ---------------------------------------------------------------------------
__global__ __launch_bounds__(256) void layernorm_kernel(
    const float* __restrict__ y, const float* __restrict__ g, const float* __restrict__ b,
    float* __restrict__ xo, u16* __restrict__ xb) {
  const int row = blockIdx.x;
  const int tid = threadIdx.x;
  __shared__ float red[256];
  const float* yr = y + (size_t)row * DM;
  float a0 = yr[tid], a1 = yr[tid + 256];
  red[tid] = a0 + a1; __syncthreads();
  for (int s = 128; s > 0; s >>= 1) { if (tid < s) red[tid] += red[tid + s]; __syncthreads(); }
  float mu = red[0] * (1.0f / DM); __syncthreads();
  float d0 = a0 - mu, d1 = a1 - mu;
  red[tid] = d0 * d0 + d1 * d1; __syncthreads();
  for (int s = 128; s > 0; s >>= 1) { if (tid < s) red[tid] += red[tid + s]; __syncthreads(); }
  float rstd = rsqrtf(red[0] * (1.0f / DM) + 1e-5f);
  float v0 = d0 * rstd * g[tid] + b[tid];
  float v1 = d1 * rstd * g[tid + 256] + b[tid + 256];
  float* xr = xo + (size_t)row * DM;
  xr[tid] = v0; xr[tid + 256] = v1;
  u16* xbr = xb + (size_t)row * DM;
  xbr[tid] = f2bf(v0); xbr[tid + 256] = f2bf(v1);
}

// ---------------------------------------------------------------------------
// final (s,b)->(b,s) transpose into d_out
// ---------------------------------------------------------------------------
__global__ __launch_bounds__(256) void untranspose_kernel(
    const float* __restrict__ x, float* __restrict__ out) {
  const int blk = blockIdx.x;       // b*S + s
  const int b = blk >> 8;
  const int s = blk & 255;
  const int irow = s * Bx + b;
  for (int c = threadIdx.x; c < DM; c += 256)
    out[(size_t)blk * DM + c] = x[(size_t)irow * DM + c];
}

// ---------------------------------------------------------------------------
static void launch_gemm(hipStream_t stream,
    const u16* A, long long lda, long long aoo, long long aoi,
    const u16* Bm, long long ldb, long long boo, long long boi,
    float* Cf, u16* Cb, long long ldc, long long coo, long long coi,
    const float* bias, const float* resid,
    int M, int N, int K, int batches, int inner_n, float scale, int relu) {
  dim3 grid(N / 32, M / 32, batches);
  wmma_gemm_bf16<<<grid, 32, 0, stream>>>(A, lda, aoo, aoi, Bm, ldb, boo, boi,
      Cf, Cb, ldc, coo, coi, bias, resid, M, N, K, inner_n, scale, relu);
}

static void launch_cvt(hipStream_t stream, const float* in, u16* out, long long n) {
  int blocks = (int)((n + 255) / 256);
  if (blocks > 4096) blocks = 4096;
  f32_to_bf16_kernel<<<blocks, 256, 0, stream>>>(in, out, n);
}

extern "C" void kernel_launch(void* const* d_in, const int* in_sizes, int n_in,
                              void* d_out, int out_size, void* d_ws, size_t ws_size,
                              hipStream_t stream) {
  const float* lm   = (const float*)d_in[0];
  const int*   eidx = (const int*)d_in[1];
  const int E = in_sizes[1] / 2;
  const float* g1w  = (const float*)d_in[2];
  const float* g1as = (const float*)d_in[3];
  const float* g1ad = (const float*)d_in[4];
  const float* g1b  = (const float*)d_in[5];
  const float* g2w  = (const float*)d_in[6];
  const float* g2as = (const float*)d_in[7];
  const float* g2ad = (const float*)d_in[8];
  const float* g2b  = (const float*)d_in[9];
  const float* fcw  = (const float*)d_in[10];
  const float* fcb  = (const float*)d_in[11];
  const float* pe   = (const float*)d_in[12];
  const float* ipw  = (const float*)d_in[13];
  const float* ipb  = (const float*)d_in[14];
  const float* opw  = (const float*)d_in[15];
  const float* opb  = (const float*)d_in[16];
  const float* l1g  = (const float*)d_in[17];
  const float* l1b  = (const float*)d_in[18];
  const float* fw1  = (const float*)d_in[19];
  const float* fb1  = (const float*)d_in[20];
  const float* fw2  = (const float*)d_in[21];
  const float* fb2  = (const float*)d_in[22];
  const float* l2g  = (const float*)d_in[23];
  const float* l2b  = (const float*)d_in[24];

  char* p = (char*)d_ws;
  auto alloc = [&](size_t bytes) -> char* {
    char* r = p;
    p += (bytes + 255) & ~(size_t)255;
    return r;
  };

  u16*   gat_bf  = (u16*)alloc((size_t)TOK * (Lnodes * 64) * 2);        // [G, 4352]
  u16*   fcw_bf  = (u16*)alloc((size_t)DM * (Lnodes * 64) * 2);
  u16*   ipw_bf  = (u16*)alloc((size_t)NL * 3 * DM * DM * 2);
  u16*   opw_bf  = (u16*)alloc((size_t)NL * DM * DM * 2);
  u16*   fw1_bf  = (u16*)alloc((size_t)NL * DFF * DM * 2);
  u16*   fw2_bf  = (u16*)alloc((size_t)NL * DM * DFF * 2);
  float* fc_out  = (float*)alloc((size_t)TOK * DM * 4);
  float* xf      = (float*)alloc((size_t)TOK * DM * 4);                 // token layout s*B+b
  u16*   x_bf    = (u16*)alloc((size_t)TOK * DM * 2);
  u16*   qkv_bf  = (u16*)alloc((size_t)TOK * 3 * DM * 2);
  u16*   vt_bf   = (u16*)alloc((size_t)Bx * NH * DH * Sx * 2);
  float* scores  = (float*)alloc((size_t)Bx * NH * Sx * Sx * 4);
  u16*   attn_bf = (u16*)alloc((size_t)Bx * NH * Sx * Sx * 2);
  u16*   obuf_bf = (u16*)alloc((size_t)TOK * DM * 2);
  u16*   ff_bf   = (u16*)alloc((size_t)TOK * DFF * 2);
  float* yf      = (float*)alloc((size_t)TOK * DM * 4);
  (void)ws_size; (void)n_in; (void)out_size;

  // weight conversions to bf16
  launch_cvt(stream, fcw, fcw_bf, (long long)DM * (Lnodes * 64));
  launch_cvt(stream, ipw, ipw_bf, (long long)NL * 3 * DM * DM);
  launch_cvt(stream, opw, opw_bf, (long long)NL * DM * DM);
  launch_cvt(stream, fw1, fw1_bf, (long long)NL * DFF * DM);
  launch_cvt(stream, fw2, fw2_bf, (long long)NL * DM * DFF);

  // GAT frontend
  gat_kernel<<<TOK, 128, 0, stream>>>(lm, eidx, E, g1w, g1as, g1ad, g1b,
                                      g2w, g2as, g2ad, g2b, gat_bf);

  // FC: [2048, 4352] x [512, 4352]^T -> fp32 [2048, 512] (rows g = b*S+s)
  launch_gemm(stream, gat_bf, Lnodes * 64, 0, 0,
              fcw_bf, Lnodes * 64, 0, 0,
              fc_out, nullptr, DM, 0, 0, fcb, nullptr,
              TOK, DM, Lnodes * 64, 1, 1, 1.0f, 0);

  // transpose to seq-first + positional encoding
  posenc_kernel<<<TOK, 256, 0, stream>>>(fc_out, pe, xf, x_bf);

  const long long ldq = (long long)Bx * 3 * DM;   // 12288: qkv row stride over s
  for (int l = 0; l < NL; ++l) {
    // QKV projection -> bf16 [TOK, 1536]
    launch_gemm(stream, x_bf, DM, 0, 0,
                ipw_bf + (size_t)l * 3 * DM * DM, DM, 0, 0,
                nullptr, qkv_bf, 3 * DM, 0, 0, ipb + (size_t)l * 3 * DM, nullptr,
                TOK, 3 * DM, DM, 1, 1, 1.0f, 0);

    // scores[b,h] = (Q K^T) / 8 : batched over 64 (b,h); K stored [t,d] = [N,K]
    launch_gemm(stream, qkv_bf, ldq, 3 * DM, DH,
                qkv_bf + DM, ldq, 3 * DM, DH,
                scores, nullptr, Sx, (long long)NH * Sx * Sx, (long long)Sx * Sx,
                nullptr, nullptr, Sx, Sx, DH, Bx * NH, NH, 0.125f, 0);

    softmax_kernel<<<Bx * NH * Sx, 256, 0, stream>>>(scores, attn_bf);

    // V^T so attn@V also hits the [N,K] contiguous path
    transpose_v_kernel<<<Bx * NH, 256, 0, stream>>>(qkv_bf, vt_bf);

    // O[b,h] = attn V -> bf16 [TOK, 512] (head-sliced columns)
    launch_gemm(stream, attn_bf, Sx, (long long)NH * Sx * Sx, (long long)Sx * Sx,
                vt_bf, Sx, (long long)NH * DH * Sx, (long long)DH * Sx,
                nullptr, obuf_bf, (long long)Bx * DM, DM, DH,
                nullptr, nullptr, Sx, DH, Sx, Bx * NH, NH, 1.0f, 0);

    // out projection + residual -> yf fp32
    launch_gemm(stream, obuf_bf, DM, 0, 0,
                opw_bf + (size_t)l * DM * DM, DM, 0, 0,
                yf, nullptr, DM, 0, 0, opb + (size_t)l * DM, xf,
                TOK, DM, DM, 1, 1, 1.0f, 0);

    layernorm_kernel<<<TOK, 256, 0, stream>>>(yf, l1g + (size_t)l * DM, l1b + (size_t)l * DM, xf, x_bf);

    // FFN up + relu -> bf16 [TOK, 2048]
    launch_gemm(stream, x_bf, DM, 0, 0,
                fw1_bf + (size_t)l * DFF * DM, DM, 0, 0,
                nullptr, ff_bf, DFF, 0, 0, fb1 + (size_t)l * DFF, nullptr,
                TOK, DFF, DM, 1, 1, 1.0f, 1);

    // FFN down + residual -> yf fp32
    launch_gemm(stream, ff_bf, DFF, 0, 0,
                fw2_bf + (size_t)l * DM * DFF, DFF, 0, 0,
                yf, nullptr, DM, 0, 0, fb2 + (size_t)l * DM, xf,
                TOK, DM, DFF, 1, 1, 1.0f, 0);

    layernorm_kernel<<<TOK, 256, 0, stream>>>(yf, l2g + (size_t)l * DM, l2b + (size_t)l * DM, xf, x_bf);
  }

  untranspose_kernel<<<TOK, 256, 0, stream>>>(xf, (float*)d_out);
}
